// CycleGNN_58901181498080
// MI455X (gfx1250) — matrix operand order repaired and checked by reading, not compiled, not run on hardware
//
#include <hip/hip_runtime.h>

// ---------------------------------------------------------------------------
// CycleGNN on gfx1250 (MI455X, wave32).
//   B=256 graphs x NPER=512 nodes (contiguous segments), D=32, F=256,
//   E=524288 edges, 4 outer steps.
// Edge GEMM [E,32]@[32,32] runs on v_wmma_f32_16x16x32_f16 (f32 accumulate),
// W_msg pre-packed once into the per-lane WMMA B layout; scatter-add via
// hardware global_atomic_add_f32. Per-graph reductions / projection /
// line-search are block-per-graph with LDS reductions (proj fits the 192MB L2).
// ---------------------------------------------------------------------------

#define BGR   256
#define NPER  512
#define NN    (BGR * NPER)      // 131072 nodes
#define FD    256
#define DD    32
#define EE    524288
#define NSTEP 4

typedef __attribute__((ext_vector_type(16))) _Float16 v16h;
typedef __attribute__((ext_vector_type(8)))  _Float16 v8h;
typedef __attribute__((ext_vector_type(8)))  float    v8f;

// ---- init: xs = x_start ---------------------------------------------------
__global__ __launch_bounds__(256) void k_init_xs(const float* __restrict__ x0,
                                                 float* __restrict__ xs) {
  int i = blockIdx.x * 256 + threadIdx.x;
  if (i < NN) xs[i] = x0[i];
}

// ---- pre-pack W_msg (f32 row-major [K][N]) into WMMA B-operand layout -----
// B packing (contiguous k, same order as A in k_edges): lane l<16 holds col N=l, K=0..15 in halves 0..15;
// lane l>=16 holds col N=l-16, K=16..31.  Buffer: [colhalf][lane][16 halves].
__global__ void k_pack_wmsg(const float* __restrict__ Wmsg,
                            _Float16* __restrict__ Wp) {
  int t = threadIdx.x;                 // 64 threads: (colhalf, lane)
  int ch = t >> 5, lane = t & 31;
  int half = lane >> 4, l16 = lane & 15;
  int ncol = ch * 16 + l16;
  for (int j = 0; j < 16; ++j)         // K = half*16 + j
    Wp[(ch * 32 + lane) * 16 + j] = (_Float16)Wmsg[(half * 16 + j) * DD + ncol];
}

// ---- zero agg buffer (float4 stores) --------------------------------------
__global__ __launch_bounds__(256) void k_zero_agg(float4* __restrict__ agg4) {
  int i = blockIdx.x * 256 + threadIdx.x;   // NN*DD/4 threads
  agg4[i] = make_float4(0.f, 0.f, 0.f, 0.f);
}

// ---- node embedding: h = relu([xs,q] @ W_in + b_in); also f16 copy --------
// one wave per node, lane = feature d (fully coalesced 32-lane rows)
__global__ __launch_bounds__(256) void k_embed(const float* __restrict__ xs,
                                               const float* __restrict__ q,
                                               const float* __restrict__ Win,
                                               const float* __restrict__ bin,
                                               float* __restrict__ hf,
                                               _Float16* __restrict__ h16) {
  int t = blockIdx.x * 256 + threadIdx.x;   // NN*DD threads
  int n = t >> 5;
  int d = t & 31;
  float v = fmaf(xs[n], Win[d], fmaf(q[n], Win[DD + d], bin[d]));
  v = v > 0.f ? v : 0.f;
  hf[t]  = v;
  h16[t] = (_Float16)v;
}

// ---- edge GEMM + scatter: agg[dst] += h[src] @ W_msg + b_msg --------------
// One wave handles 32 edges = 2 tiles of 16; B/bias loaded once per wave.
// A layout (contiguous k, same as packed B): lane l<16 holds row l, halves
// [K0..15]; lane l>=16 holds row l-16, halves [K16..31].
// C layout: lane holds col (l&15); VGPR v holds row v (+8 for lanes 16..31).
__global__ __launch_bounds__(256) void k_edges(const _Float16* __restrict__ h16,
                                               const _Float16* __restrict__ Wp,
                                               const float* __restrict__ bmsg,
                                               const int* __restrict__ esrc,
                                               const int* __restrict__ edst,
                                               float* __restrict__ agg) {
  int wave = (blockIdx.x * 256 + threadIdx.x) >> 5;  // EE/32 waves total
  int lane = threadIdx.x & 31;
  int half = lane >> 4;        // 0: lanes 0-15, 1: lanes 16-31
  int l16  = lane & 15;
  int kb    = half * 16;       // A K-offset for this lane
  int mbase = half * 8;        // C row base for this lane

  // B operands + bias: loaded once, reused across both edge tiles
  v16h b0 = *(const v16h*)(Wp + (0 * 32 + lane) * 16);   // out cols 0..15
  v16h b1 = *(const v16h*)(Wp + (1 * 32 + lane) * 16);   // out cols 16..31
  float bb0 = bmsg[l16];
  float bb1 = bmsg[16 + l16];

  int ebase = wave * 32;
#pragma unroll
  for (int tile = 0; tile < 2; ++tile) {
    int e0 = ebase + tile * 16;

    // A: gathered h row for edge e0+l16 (two aligned 16B chunks)
    int src = esrc[e0 + l16];
    v8h lo = *(const v8h*)(h16 + (size_t)src * DD + kb);        // K kb..kb+7
    v8h hi = *(const v8h*)(h16 + (size_t)src * DD + kb + 8);    // K kb+8..+15
    v16h a;
#pragma unroll
    for (int i = 0; i < 8; ++i) { a[i] = lo[i]; a[8 + i] = hi[i]; }

    // destination node ids for the 8 C rows this lane owns
    int dsts[8];
#pragma unroll
    for (int v = 0; v < 8; ++v) dsts[v] = edst[e0 + mbase + v];

    v8f c0 = {}, c1 = {};
    c0 = __builtin_amdgcn_wmma_f32_16x16x32_f16(false, a, false, b0,
                                                (short)0, c0, false, false);
    c1 = __builtin_amdgcn_wmma_f32_16x16x32_f16(false, a, false, b1,
                                                (short)0, c1, false, false);
#pragma unroll
    for (int v = 0; v < 8; ++v) {
      size_t rb = (size_t)dsts[v] * DD;
      unsafeAtomicAdd(&agg[rb + l16],      c0[v] + bb0);
      unsafeAtomicAdd(&agg[rb + 16 + l16], c1[v] + bb1);
    }
  }
}

// ---- per-graph: pred, labels, direction -----------------------------------
// block b = graph b (512 threads, one per node). LDS reductions for the two
// per-graph L1 sums (|x_sol - xs| and |pred|).
__global__ __launch_bounds__(512) void k_nodeout(const float* __restrict__ hf,
                                                 const float* __restrict__ agg,
                                                 const float* __restrict__ xs,
                                                 const float* __restrict__ xsol,
                                                 const float* __restrict__ Wout,
                                                 const float* __restrict__ bout,
                                                 float* __restrict__ out,
                                                 float* __restrict__ dd,
                                                 float tau, int step) {
  __shared__ float s1[NPER];
  __shared__ float s2[NPER];
  int b = blockIdx.x, t = threadIdx.x;
  int n = b * NPER + t;

  float pred = bout[0];
#pragma unroll
  for (int d = 0; d < DD; ++d) {
    float v = hf[(size_t)n * DD + d] + agg[(size_t)n * DD + d];
    v = v > 0.f ? v : 0.f;               // h2 = relu(h + agg)
    pred = fmaf(v, Wout[d], pred);
  }
  float x = xs[n];
  float r = xsol[n] - x;
  s1[t] = fabsf(r);
  s2[t] = fabsf(pred);
  __syncthreads();
  for (int off = 256; off > 0; off >>= 1) {
    if (t < off) { s1[t] += s1[t + off]; s2[t] += s2[t + off]; }
    __syncthreads();
  }
  float sr = s1[0] + 1e-12f;
  float sp = s2[0] + 1e-12f;

  out[(size_t)n * NSTEP + step] = pred;                         // preds[n,step]
  out[(size_t)NN * NSTEP + (size_t)n * NSTEP + step] = r / sr;  // labels
  dd[n] = pred / sp + 3.0f * tau / (x + tau);                   // direction
}

// ---- per-graph: nullspace projection + line search + xs update ------------
// df = proj[b]^T dd (256), pproj = proj[b] df (512), alpha = per-graph min
// ratio clipped to [0,5] * 0.995, xs += alpha * pproj.
__global__ __launch_bounds__(256) void k_project(const float* __restrict__ proj,
                                                 const float* __restrict__ dd,
                                                 float* __restrict__ xs) {
  __shared__ float s_dd[NPER];
  __shared__ float s_df[FD];
  __shared__ float s_pp[NPER];
  __shared__ float s_red[256];
  int b = blockIdx.x, t = threadIdx.x;
  const float* P = proj + (size_t)b * NPER * FD;

  s_dd[t]       = dd[b * NPER + t];
  s_dd[t + 256] = dd[b * NPER + t + 256];
  __syncthreads();

  // df[f]: thread f streams column f (coalesced across threads per row)
  float acc = 0.f;
  for (int n = 0; n < NPER; ++n)
    acc = fmaf(P[(size_t)n * FD + t], s_dd[n], acc);
  s_df[t] = acc;
  __syncthreads();

  // pproj[n]: one wave per row, shfl-xor reduction within wave32
  int wave = t >> 5, lane = t & 31;
  for (int i = 0; i < 64; ++i) {
    int n = wave * 64 + i;
    float part = 0.f;
#pragma unroll
    for (int j = 0; j < 8; ++j)
      part = fmaf(P[(size_t)n * FD + lane + 32 * j], s_df[lane + 32 * j], part);
#pragma unroll
    for (int sh = 16; sh > 0; sh >>= 1)
      part += __shfl_xor(part, sh, 32);
    if (lane == 0) s_pp[n] = part;
  }
  __syncthreads();

  // line search: per-graph min ratio
  float rmin = 5.0f;
#pragma unroll
  for (int k = 0; k < 2; ++k) {
    int n = t + k * 256;
    float d = s_pp[n];
    float x = xs[b * NPER + n];
    float ratio = (d < 0.f) ? (x / fmaxf(-d, 1e-12f)) : 5.0f;
    rmin = fminf(rmin, ratio);
  }
  s_red[t] = rmin;
  __syncthreads();
  for (int off = 128; off > 0; off >>= 1) {
    if (t < off) s_red[t] = fminf(s_red[t], s_red[t + off]);
    __syncthreads();
  }
  float alpha = fminf(fmaxf(s_red[0], 0.f), 5.0f) * 0.995f;

#pragma unroll
  for (int k = 0; k < 2; ++k) {
    int n = t + k * 256;
    xs[b * NPER + n] += alpha * s_pp[n];
  }
}

// ---------------------------------------------------------------------------
extern "C" void kernel_launch(void* const* d_in, const int* in_sizes, int n_in,
                              void* d_out, int out_size, void* d_ws, size_t ws_size,
                              hipStream_t stream) {
  const float* x_start = (const float*)d_in[0];
  const float* x_sol   = (const float*)d_in[1];
  const float* q       = (const float*)d_in[2];
  const float* proj    = (const float*)d_in[3];
  const float* W_in    = (const float*)d_in[4];
  const float* b_in    = (const float*)d_in[5];
  const float* W_msg   = (const float*)d_in[6];
  const float* b_msg   = (const float*)d_in[7];
  const float* W_out   = (const float*)d_in[8];
  const float* b_out   = (const float*)d_in[9];
  const int*   esrc    = (const int*)d_in[10];
  const int*   edst    = (const int*)d_in[11];
  // d_in[12] (vals_batch) unused: segments are contiguous NPER blocks.

  float* out = (float*)d_out;

  // workspace layout (~41MB)
  char* w = (char*)d_ws;
  float*    xs  = (float*)w;     w += (size_t)NN * 4;           // iterate
  float*    hf  = (float*)w;     w += (size_t)NN * DD * 4;      // h f32
  _Float16* h16 = (_Float16*)w;  w += (size_t)NN * DD * 2;      // h f16 (WMMA A)
  float*    agg = (float*)w;     w += (size_t)NN * DD * 4;      // segment_sum
  float*    dd  = (float*)w;     w += (size_t)NN * 4;           // direction
  _Float16* Wp  = (_Float16*)w;  w += (size_t)2 * 32 * 16 * 2;  // packed W_msg

  k_init_xs<<<(NN + 255) / 256, 256, 0, stream>>>(x_start, xs);
  k_pack_wmsg<<<1, 64, 0, stream>>>(W_msg, Wp);

  const float taus[NSTEP] = {0.1f, 0.05f, 0.025f, 0.0125f};
  for (int step = 0; step < NSTEP; ++step) {
    k_zero_agg<<<(NN * DD / 4) / 256, 256, 0, stream>>>((float4*)agg);
    k_embed<<<(NN * DD) / 256, 256, 0, stream>>>(xs, q, W_in, b_in, hf, h16);
    // EE/32 waves (2 tiles of 16 edges each), 8 waves per 256-thread block
    k_edges<<<EE / (32 * 8), 256, 0, stream>>>(h16, Wp, b_msg, esrc, edst, agg);
    k_nodeout<<<BGR, 512, 0, stream>>>(hf, agg, xs, x_sol, W_out, b_out,
                                       out, dd, taus[step], step);
    k_project<<<BGR, 256, 0, stream>>>(proj, dd, xs);
  }
}
